// InterferenceLayer_9543417331819
// MI455X (gfx1250) — matrix operand order repaired and checked by reading, not compiled, not run on hardware
//
#include <hip/hip_runtime.h>
#include <hip/hip_bf16.h>
#include <math.h>

// ---------------- problem constants (match reference) ----------------
#define TT   2048
#define CC   512
#define HH   8
#define HD   64
#define FFN  2048
#define JJ   52
#define QKVN (3*CC)          // 1536

__device__ __constant__ int d_offsets[JJ] = {
  0,1,2,3,4,5,6,7,8,9,10,11,12,13,14,15,16,17,18,19,20,
  21,22,23,24,25,26,27,28,29,30,31,32,33,34,35,36,37,38,39,40,
  96,128,145,163,185,209,236,266,301,340,384
};

// ---------------- types ----------------
typedef __attribute__((ext_vector_type(16))) __bf16 v16bf;
typedef __attribute__((ext_vector_type(8)))  float  v8f;

union FragBF { v16bf v; unsigned int u[8]; };

static __device__ inline __bf16 f2b(float f) {
  union { float f; unsigned u; } x; x.f = f;
  unsigned r = (x.u + 0x7FFFu + ((x.u >> 16) & 1u)) >> 16;
  union { unsigned short s; __bf16 b; } y; y.s = (unsigned short)r;
  return y.b;
}

// ---------------- fp32 -> bf16 weight conversion ----------------
__global__ void cvt_bf16_kernel(const float* __restrict__ src, __bf16* __restrict__ dst, int n) {
  int i = blockIdx.x * blockDim.x + threadIdx.x;
  if (i < n) dst[i] = f2b(src[i]);
}

// ---------------- layernorm (C=512) -> bf16, one block per row ----------------
__global__ void ln_bf16_kernel(const float* __restrict__ x, const float* __restrict__ w,
                               const float* __restrict__ b, __bf16* __restrict__ out) {
  const int t = blockIdx.x, tid = threadIdx.x;
  const float* row = x + (size_t)t * CC;
  float v0 = row[tid], v1 = row[tid + 256];
  __shared__ float s1[256], s2[256];
  s1[tid] = v0 + v1; s2[tid] = v0 * v0 + v1 * v1;
  __syncthreads();
  for (int st = 128; st > 0; st >>= 1) {
    if (tid < st) { s1[tid] += s1[tid + st]; s2[tid] += s2[tid + st]; }
    __syncthreads();
  }
  float m = s1[0] * (1.0f / CC);
  float var = s2[0] * (1.0f / CC) - m * m;
  float r = rsqrtf(var + 1e-5f);
  out[(size_t)t * CC + tid]       = f2b((v0 - m) * r * w[tid]       + b[tid]);
  out[(size_t)t * CC + tid + 256] = f2b((v1 - m) * r * w[tid + 256] + b[tid + 256]);
}

// ---------------- bf16 WMMA GEMM: out[M,N] = A[M,K] @ W[N,K]^T ----------------
// block = 128 threads (4 wave32), tile 128(M) x 64(N); each wave owns 32x64
// (8 accumulators, 8 WMMA per K-step of 32). Double-buffered LDS tiles,
// one barrier per K-step; global loads via base+int-offset (keeps GLOBAL path).
#define LSTR 40   // padded LDS row stride in shorts (80B, 16B-aligned rows)
__global__ __launch_bounds__(128)
void gemm_bf16_kernel(const __bf16* __restrict__ A, const __bf16* __restrict__ W,
                      const float* __restrict__ bias, const float* __restrict__ add,
                      float* __restrict__ outF, __bf16* __restrict__ outH,
                      int M, int N, int K, int act) {
  __shared__ unsigned short sA[2][128 * LSTR];
  __shared__ unsigned short sB[2][64 * LSTR];
  const int tid  = threadIdx.x;
  const int lane = tid & 31, wave = tid >> 5;     // 4 waves
  const int g = lane >> 4, lm = lane & 15;
  const int m0 = blockIdx.y * 128, n0 = blockIdx.x * 64;

  v8f c[2][4];
#pragma unroll
  for (int mt = 0; mt < 2; ++mt)
#pragma unroll
    for (int j = 0; j < 4; ++j) c[mt][j] = (v8f){};

  // cooperative-load geometry: A tile 128x32 = 512 uint4 (4/thread),
  // B tile 64x32 = 256 uint4 (2/thread); integer element offsets from A/W base.
  int aBase[4], aO[4];
#pragma unroll
  for (int i = 0; i < 4; ++i) {
    int e = tid + 128 * i, r = e >> 2, q = e & 3;
    aBase[i] = (m0 + r) * K + q * 8;
    aO[i]    = r * LSTR + q * 8;
  }
  int bBase[2], bO[2];
#pragma unroll
  for (int i = 0; i < 2; ++i) {
    int e = tid + 128 * i, r = e >> 2, q = e & 3;
    bBase[i] = (n0 + r) * K + q * 8;
    bO[i]    = r * LSTR + q * 8;
  }

  const int nk = K >> 5;
  for (int kt = 0; kt < nk; ++kt) {
    const int koff = kt << 5;
    const int buf  = kt & 1;

    // batch global loads into registers (overlap with previous step's WMMAs)
    uint4 da[4], db[2];
#pragma unroll
    for (int i = 0; i < 4; ++i) {
      const __bf16* p = A + aBase[i] + koff;
      da[i] = *(const uint4*)p;
      __builtin_prefetch(p + 32, 0, 1);           // global_prefetch_b8 (speculative)
    }
#pragma unroll
    for (int i = 0; i < 2; ++i) {
      const __bf16* p = W + bBase[i] + koff;
      db[i] = *(const uint4*)p;
      __builtin_prefetch(p + 32, 0, 1);
    }
    // stage into this step's LDS buffer (previous barrier freed it)
#pragma unroll
    for (int i = 0; i < 4; ++i) *(uint4*)&sA[buf][aO[i]] = da[i];
#pragma unroll
    for (int i = 0; i < 2; ++i) *(uint4*)&sB[buf][bO[i]] = db[i];
    __syncthreads();

    // A fragments (16x32 bf16), ISA layout: k = 16*(v/4) + 8*g + 2*(v&3) + p
    FragBF fa[2];
#pragma unroll
    for (int mt = 0; mt < 2; ++mt) {
      const int aRow = (wave * 32 + mt * 16 + lm) * LSTR;
#pragma unroll
      for (int v = 0; v < 8; ++v) {
        int k = ((v >> 2) << 4) + (g << 3) + ((v & 3) << 1);
        fa[mt].u[v] = *(const unsigned int*)&sA[buf][aRow + k];
      }
    }
    // B fragments (32x16 bf16), ISA layout: k = 16*(v/4) + 8*g + 2*(v&3) + p
#pragma unroll
    for (int j = 0; j < 4; ++j) {
      FragBF fb;
      const int bRow = (j * 16 + lm) * LSTR;
#pragma unroll
      for (int v = 0; v < 8; ++v) {
        int k = ((v >> 2) << 4) + (g << 3) + ((v & 3) << 1);
        fb.u[v] = *(const unsigned int*)&sB[buf][bRow + k];
      }
      c[0][j] = __builtin_amdgcn_wmma_f32_16x16x32_bf16(
          false, fa[0].v, false, fb.v, (short)0, c[0][j], false, false);
      c[1][j] = __builtin_amdgcn_wmma_f32_16x16x32_bf16(
          false, fa[1].v, false, fb.v, (short)0, c[1][j], false, false);
    }
  }

  // epilogue: C/D layout m = r + 8*g, n = lane&15
#pragma unroll
  for (int mt = 0; mt < 2; ++mt) {
#pragma unroll
    for (int j = 0; j < 4; ++j) {
#pragma unroll
      for (int r = 0; r < 8; ++r) {
        int row = m0 + wave * 32 + mt * 16 + r + 8 * g;
        int col = n0 + j * 16 + lm;
        float v = c[mt][j][r];
        if (bias) v += bias[col];
        if (add)  v += add[(size_t)row * N + col];
        if (act == 1) v = 0.5f * v * (1.0f + erff(v * 0.70710678118654752f));  // exact GELU
        size_t o = (size_t)row * N + col;
        if (outF) outF[o] = v;
        if (outH) outH[o] = f2b(v);
      }
    }
  }
}

// ---------------- sparse-offset attention: one 64-thread block per (t, h) ----------------
__global__ void attn_kernel(const float* __restrict__ qkv, const float* __restrict__ pos_bias,
                            const float* __restrict__ scale_embed, const float* __restrict__ if_gain,
                            __bf16* __restrict__ ctx) {
  const int t = blockIdx.x, h = blockIdx.y, tid = threadIdx.x;
  __shared__ float qs[64], ss[64];
  __shared__ int   gidx[64];
  __shared__ float mx_s, sum_s;

  qs[tid] = qkv[(size_t)t * QKVN + h * HD + tid];
  __syncthreads();

  if (tid < JJ) {
    int off = d_offsets[tid];
    int idx = t - off;
    bool valid = (idx >= 0);
    int idxc = valid ? idx : 0;
    gidx[tid] = idxc;
    const float* kg = qkv + (size_t)idxc * QKVN + CC + h * HD;
    const float* se = scale_embed + tid * HD;
    float s = 0.f;
#pragma unroll 8
    for (int d = 0; d < HD; ++d) s += qs[d] * (kg[d] + se[d]);
    s *= 0.125f;  // 1/sqrt(64)
    bool distal = (off > 350);
    bool local  = (h < 7);
    float pb = pos_bias[tid * HH + h];
    if (distal && local)   pb = -10000.0f;
    if (!distal && !local) pb = -3.0f;
    s += pb;
    if (!valid) s = -1e9f;
    ss[tid] = s;
  } else { ss[tid] = -1e30f; gidx[tid] = 0; }
  __syncthreads();
  if (tid == 0) { float m = -1e30f; for (int j = 0; j < JJ; ++j) m = fmaxf(m, ss[j]); mx_s = m; }
  __syncthreads();
  if (tid < JJ) ss[tid] = __expf(ss[tid] - mx_s);
  __syncthreads();
  if (tid == 0) { float s = 0.f; for (int j = 0; j < JJ; ++j) s += ss[j]; sum_s = s; }
  __syncthreads();

  float inv = 1.0f / sum_s;
  float acc = 0.f;
  for (int j = 0; j < JJ; ++j)
    acc += ss[j] * qkv[(size_t)gidx[j] * QKVN + 2 * CC + h * HD + tid];
  acc *= inv * if_gain[h];
  ctx[(size_t)t * CC + h * HD + tid] = f2b(acc);
}

// ---------------- roll-multiply + AGC, one block per row t ----------------
__global__ void roll_agc_kernel(const float* __restrict__ x1, const float* __restrict__ alpha,
                                const float* __restrict__ beta, float* __restrict__ x3) {
  const int t = blockIdx.x, tid = threadIdx.x;
  const int tm1 = (t == 0) ? (TT - 1) : (t - 1);
  const float ta = tanhf(alpha[0]), tb = tanhf(beta[0]);
  float a0 = x1[(size_t)t * CC + tid],        a1 = x1[(size_t)t * CC + tid + 256];
  float r0 = x1[(size_t)tm1 * CC + tid],      r1 = x1[(size_t)tm1 * CC + tid + 256];
  float y0 = a0 + ta * a0 * r0, y1 = a1 + ta * a1 * r1;
  __shared__ float s[256];
  s[tid] = y0 * y0 + y1 * y1;
  __syncthreads();
  for (int st = 128; st > 0; st >>= 1) { if (tid < st) s[tid] += s[tid + st]; __syncthreads(); }
  float nrm = fmaxf(sqrtf(s[0]), 1e-6f);
  float sc = tb / nrm;
  x3[(size_t)t * CC + tid]       = y0 + sc * y0;
  x3[(size_t)t * CC + tid + 256] = y1 + sc * y1;
}

// ---------------- causal 256-tap geometric EMA, thread per (t,c) ----------------
__global__ void ema_kernel(const float* __restrict__ x3, const float* __restrict__ ema_alpha,
                           const float* __restrict__ ema_gain, const float* __restrict__ gamma,
                           float* __restrict__ x4) {
  int gid = blockIdx.x * blockDim.x + threadIdx.x;   // 0 .. T*C-1
  int t = gid >> 9, c = gid & (CC - 1);
  float a = 1.0f / (1.0f + __expf(-ema_alpha[c]));
  a = fminf(fmaxf(a, 0.01f), 0.99f);
  float denom = (1.0f - powf(a, 256.0f)) / (1.0f - a);
  int kl = (t + 1 < 256) ? (t + 1) : 256;
  float acc = 0.f, w = 1.f;
  const float* col = x3 + c;
  for (int k = 0; k < kl; ++k) { acc += w * col[(size_t)(t - k) << 9]; w *= a; }
  x4[gid] = x3[gid] + tanhf(gamma[0]) * (acc / denom) * ema_gain[c];
}

// ---------------- launch ----------------
extern "C" void kernel_launch(void* const* d_in, const int* in_sizes, int n_in,
                              void* d_out, int out_size, void* d_ws, size_t ws_size,
                              hipStream_t stream) {
  const float* x          = (const float*)d_in[0];
  const float* qkv_w      = (const float*)d_in[1];
  const float* out_w      = (const float*)d_in[2];
  const float* pos_bias   = (const float*)d_in[3];
  const float* scale_emb  = (const float*)d_in[4];
  const float* if_gain    = (const float*)d_in[5];
  const float* ln1_w      = (const float*)d_in[6];
  const float* ln1_b      = (const float*)d_in[7];
  const float* ln2_w      = (const float*)d_in[8];
  const float* ln2_b      = (const float*)d_in[9];
  const float* ffn_w1     = (const float*)d_in[10];
  const float* ffn_b1     = (const float*)d_in[11];
  const float* ffn_w2     = (const float*)d_in[12];
  const float* ffn_b2     = (const float*)d_in[13];
  const float* ema_alpha  = (const float*)d_in[14];
  const float* ema_gain   = (const float*)d_in[15];
  const float* alpha      = (const float*)d_in[16];
  const float* beta       = (const float*)d_in[17];
  const float* gamma      = (const float*)d_in[18];
  float* out = (float*)d_out;

  // workspace carve-up (~46 MB)
  char* ws = (char*)d_ws;
  size_t off = 0;
  auto carve = [&](size_t bytes) { void* p = ws + off; off = (off + bytes + 255) & ~(size_t)255; return p; };
  __bf16* resH   = (__bf16*)carve((size_t)TT * CC * 2);
  __bf16* qkvwH  = (__bf16*)carve((size_t)QKVN * CC * 2);
  __bf16* outwH  = (__bf16*)carve((size_t)CC * CC * 2);
  __bf16* w1H    = (__bf16*)carve((size_t)FFN * CC * 2);
  __bf16* w2H    = (__bf16*)carve((size_t)CC * FFN * 2);
  float*  qkvO   = (float*)carve((size_t)TT * QKVN * 4);
  __bf16* ctxH   = (__bf16*)carve((size_t)TT * CC * 2);
  float*  x1     = (float*)carve((size_t)TT * CC * 4);
  float*  x3     = (float*)carve((size_t)TT * CC * 4);
  float*  x4     = (float*)carve((size_t)TT * CC * 4);
  __bf16* hH     = (__bf16*)carve((size_t)TT * CC * 2);
  __bf16* gH     = (__bf16*)carve((size_t)TT * FFN * 2);
  (void)ws_size; (void)in_sizes; (void)n_in; (void)out_size;

  // weight conversion fp32 -> bf16
  cvt_bf16_kernel<<<(QKVN * CC + 255) / 256, 256, 0, stream>>>(qkv_w, qkvwH, QKVN * CC);
  cvt_bf16_kernel<<<(CC * CC + 255) / 256, 256, 0, stream>>>(out_w, outwH, CC * CC);
  cvt_bf16_kernel<<<(FFN * CC + 255) / 256, 256, 0, stream>>>(ffn_w1, w1H, FFN * CC);
  cvt_bf16_kernel<<<(CC * FFN + 255) / 256, 256, 0, stream>>>(ffn_w2, w2H, CC * FFN);

  // LN1 -> bf16 activations
  ln_bf16_kernel<<<TT, 256, 0, stream>>>(x, ln1_w, ln1_b, resH);

  // QKV GEMM: [2048,512] x [1536,512]^T -> fp32
  gemm_bf16_kernel<<<dim3(QKVN / 64, TT / 128), 128, 0, stream>>>(
      resH, qkvwH, nullptr, nullptr, qkvO, nullptr, TT, QKVN, CC, 0);

  // sparse-offset attention -> bf16 context
  attn_kernel<<<dim3(TT, HH), 64, 0, stream>>>(qkvO, pos_bias, scale_emb, if_gain, ctxH);

  // out-proj GEMM with residual x: x1 = x + ctx @ out_w^T
  gemm_bf16_kernel<<<dim3(CC / 64, TT / 128), 128, 0, stream>>>(
      ctxH, outwH, nullptr, x, x1, nullptr, TT, CC, CC, 0);

  // roll-multiply + AGC -> x3
  roll_agc_kernel<<<TT, 256, 0, stream>>>(x1, alpha, beta, x3);

  // causal geometric EMA -> x4
  ema_kernel<<<(TT * CC) / 256, 256, 0, stream>>>(x3, ema_alpha, ema_gain, gamma, x4);

  // LN2 -> bf16
  ln_bf16_kernel<<<TT, 256, 0, stream>>>(x4, ln2_w, ln2_b, hH);

  // FFN1: gelu(h @ w1^T + b1) -> bf16
  gemm_bf16_kernel<<<dim3(FFN / 64, TT / 128), 128, 0, stream>>>(
      hH, w1H, ffn_b1, nullptr, nullptr, gH, TT, FFN, CC, 1);

  // FFN2: out = x4 + g @ w2^T + b2  -> d_out (fp32)
  gemm_bf16_kernel<<<dim3(CC / 64, TT / 128), 128, 0, stream>>>(
      gH, w2H, ffn_b2, x4, out, nullptr, TT, CC, FFN, 0);
}
